// _NonLocalBlockND_45603962749103
// MI455X (gfx1250) — matrix operand order repaired and checked by reading, not compiled, not run on hardware
//
#include <hip/hip_runtime.h>
#include <cstdint>

// ---------------- problem constants ----------------
#define B_   8
#define C_   256
#define CI_  128
#define H_   64
#define W_   64
#define N_   4096   // H*W
#define M_   1024   // (H/2)*(W/2)

typedef _Float16 half_t;
typedef __attribute__((ext_vector_type(16))) _Float16 v16h;
typedef __attribute__((ext_vector_type(8)))  _Float16 v8h;
typedef __attribute__((ext_vector_type(8)))  float    v8f;

// ---------------- workspace layout (units: half elements) ----------------
// thetaT : [B][N][CI]  f16   (8 MB)
// phiT   : [B][M][CI]  f16   (2 MB)
// g      : [B][CI][M]  f16   (2 MB)
// weights: w_theta,w_phi,w_g [CI][C] f16 + w_W [C][CI] f16 (256 KB)
#define OFF_THETA ((size_t)0)
#define OFF_PHIT  ((size_t)(B_) * N_ * CI_)                 // 4194304
#define OFF_G     (OFF_PHIT + (size_t)(B_) * M_ * CI_)      // 5242880
#define OFF_WH    (OFF_G    + (size_t)(B_) * CI_ * M_)      // 6291456
// total = 6422528 halves = ~12.9 MB of ws

// -------- gfx1250 async global->LDS staging (guarded; fallback = plain copy) --
#if defined(__gfx1250__) && \
    __has_builtin(__builtin_amdgcn_global_load_async_to_lds_b128) && \
    __has_builtin(__builtin_amdgcn_s_wait_asynccnt)
#define USE_ASYNC 1
typedef int v4i_t __attribute__((vector_size(16)));
typedef __attribute__((address_space(1))) v4i_t* gas_ptr_t;
typedef __attribute__((address_space(3))) v4i_t* las_ptr_t;
// Global side: flat address == global address numerically (int->ptr cast).
// LDS side: flat LDS address truncated to 32 bits is the LDS offset
// (ISA 10.2), so int->AS3-ptr cast is exact.
#define COPY16_TO_LDS(dst, src)                                          \
  __builtin_amdgcn_global_load_async_to_lds_b128(                        \
      (gas_ptr_t)(uintptr_t)(src),                                       \
      (las_ptr_t)(uint32_t)(uintptr_t)(dst), 0, 0)
#define WAIT_ASYNC(n) __builtin_amdgcn_s_wait_asynccnt(n)
#else
#define USE_ASYNC 0
#define COPY16_TO_LDS(dst, src) (*(v8h*)(dst) = *(const v8h*)(src))
#define WAIT_ASYNC(n) ((void)0)
#endif

// ---------------- WMMA helpers (gfx1250, wave32) ----------------
static __device__ __forceinline__ v8f wmma_f16(v16h a, v16h b, v8f c) {
  // v_wmma_f32_16x16x32_f16 : D = A(16x32 f16) * B(32x16 f16) + C(16x16 f32)
  return __builtin_amdgcn_wmma_f32_16x16x32_f16(false, a, false, b, (short)0, c,
                                                false, false);
}

// A-fragment (f16 16x32): lane = M row (lp), halves: k = 16*hg + h
static __device__ __forceinline__ v16h load_a_frag(const half_t* rowp, int hg) {
  v16h a;
  ((v8h*)&a)[0] = *(const v8h*)(rowp + 16 * hg);
  ((v8h*)&a)[1] = *(const v8h*)(rowp + 16 * hg + 8);
  return a;
}

// B-fragment (f16 32x16): lane = N col (lp) + 16*hg K-half, halves k = 16*hg + h
static __device__ __forceinline__ v16h load_b_frag(const half_t* p) {
  v16h b;
  ((v8h*)&b)[0] = *(const v8h*)(p);
  ((v8h*)&b)[1] = *(const v8h*)(p + 8);
  return b;
}

// ---------------- kernel 0: convert weights to f16 once ----------------
__global__ __launch_bounds__(256) void k_cvt(const float* __restrict__ wt,
                                             const float* __restrict__ wp,
                                             const float* __restrict__ wg,
                                             const float* __restrict__ ww,
                                             half_t* __restrict__ dst) {
  int i = blockIdx.x * 256 + threadIdx.x;
  if (i < CI_ * C_) {
    dst[i]                 = (half_t)wt[i];
    dst[CI_ * C_ + i]      = (half_t)wp[i];
    dst[2 * CI_ * C_ + i]  = (half_t)wg[i];
    dst[3 * CI_ * C_ + i]  = (half_t)ww[i];  // w_W is [C][CI], same element count
  }
}

// ---------------- kernel 1: theta/phi/g 1x1 convs + fused 2x2 maxpool --------
__global__ __launch_bounds__(256) void k_proj(const float* __restrict__ x,
                                              const float* __restrict__ b_theta,
                                              const float* __restrict__ b_phi,
                                              const float* __restrict__ b_g,
                                              half_t* __restrict__ ws) {
  const int b   = blockIdx.y;
  const int ch  = blockIdx.x;       // 0..63
  const int chH = ch >> 1;          // h-pair index (pooled row)
  const int chW = ch & 1;           // which 32-wide w half
  const int tid  = threadIdx.x;
  const int lane = tid & 31, wv = tid >> 5;
  const int lp = lane & 15, hg = lane >> 4;

  // xT[n_local][c] as f16; row stride 258 halves (=129 dwords) rotates banks.
  __shared__ half_t xT[64][258];

  const size_t xbase = (size_t)b * C_ * N_;
  for (int i = 0; i < 64; ++i) {
    int idx  = i * 256 + tid;       // 0..16383 = 256 c * 64 n_local
    int c    = idx >> 6;
    int sub  = idx & 63;            // n_local = hLoc*32 + w'
    int hLoc = sub >> 5, wprime = sub & 31;
    xT[sub][c] = (half_t)x[xbase + (size_t)c * N_ +
                           (size_t)(2 * chH + hLoc) * W_ + chW * 32 + wprime];
  }
  __syncthreads();

  half_t* thetaT = ws + OFF_THETA;
  half_t* phiT   = ws + OFF_PHIT;
  half_t* gbuf   = ws + OFF_G;
  const half_t* wth = ws + OFF_WH;
  const half_t* wph = wth + CI_ * C_;
  const half_t* wgg = wph + CI_ * C_;

  const int ci0 = wv * 16;
  float bth[8], bph[8], bgg[8];
#pragma unroll
  for (int r = 0; r < 8; ++r) {
    int ci = ci0 + r + 8 * hg;      // D-tile row held by this lane in vgpr r
    bth[r] = b_theta[ci]; bph[r] = b_phi[ci]; bgg[r] = b_g[ci];
  }

  const half_t* wtrow = wth + (ci0 + lp) * C_;
  const half_t* wprow = wph + (ci0 + lp) * C_;
  const half_t* wgrow = wgg + (ci0 + lp) * C_;

  const v8f vzero = {};

#pragma unroll
  for (int jj = 0; jj < 2; ++jj) {            // two 16-wide w tiles per h-row
    v8f tA = vzero, pA = vzero, gA = vzero;   // row h=2*chH
    v8f tB = vzero, pB = vzero, gB = vzero;   // row h=2*chH+1
#pragma unroll
    for (int kk = 0; kk < 8; ++kk) {          // K = C = 256, 32 per step
      v16h aT = load_a_frag(wtrow + kk * 32, hg);
      v16h aP = load_a_frag(wprow + kk * 32, hg);
      v16h aG = load_a_frag(wgrow + kk * 32, hg);
      v16h bx0 = load_b_frag(&xT[jj * 16 + lp][kk * 32 + 16 * hg]);
      v16h bx1 = load_b_frag(&xT[32 + jj * 16 + lp][kk * 32 + 16 * hg]);
      tA = wmma_f16(aT, bx0, tA);
      pA = wmma_f16(aP, bx0, pA);
      gA = wmma_f16(aG, bx0, gA);
      tB = wmma_f16(aT, bx1, tB);
      pB = wmma_f16(aP, bx1, pB);
      gB = wmma_f16(aG, bx1, gB);
    }

    // ---- theta: + bias, store transposed [n][ci]; 8 contiguous halves/lane
    const int nA = (2 * chH) * W_ + chW * 32 + jj * 16 + lp;
    const int nB = nA + W_;
    v8h t8;
#pragma unroll
    for (int r = 0; r < 8; ++r) t8[r] = (half_t)(tA[r] + bth[r]);
    *(v8h*)(thetaT + ((size_t)b * N_ + nA) * CI_ + ci0 + 8 * hg) = t8;
#pragma unroll
    for (int r = 0; r < 8; ++r) t8[r] = (half_t)(tB[r] + bth[r]);
    *(v8h*)(thetaT + ((size_t)b * N_ + nB) * CI_ + ci0 + 8 * hg) = t8;

    // ---- phi/g: 2x2 maxpool in registers (vertical: tileA/tileB, horizontal:
    // neighbor-lane via shfl_xor 1), bias after max (max commutes with +bias).
    const int wo = chW * 16 + jj * 8 + (lp >> 1);
    const int m  = chH * 32 + wo;
#pragma unroll
    for (int r = 0; r < 8; ++r) {
      int ci = ci0 + r + 8 * hg;
      float pv = fmaxf(pA[r], pB[r]);
      pv = fmaxf(pv, __shfl_xor(pv, 1, 32));
      float gv = fmaxf(gA[r], gB[r]);
      gv = fmaxf(gv, __shfl_xor(gv, 1, 32));
      if ((lane & 1) == 0) {
        phiT[((size_t)b * M_ + m) * CI_ + ci] = (half_t)(pv + bph[r]);
        gbuf[((size_t)b * CI_ + ci) * M_ + m] = (half_t)(gv + bgg[r]);
      }
    }
  }
}

// ---------------- kernel 2: fused flash attention + w_W conv + BN + residual
// block = (tile, b); tile = 128 query rows. 8 waves, wave owns 16 n rows.
// Key chunks (32 keys) of phi/g are staged once per block into LDS, double-
// buffered via gfx1250 async global->LDS DMA (ASYNCcnt), overlapping the next
// chunk's copy with this chunk's WMMAs + online softmax.
struct StageBufs {
  half_t phi[2][32][136];   // [buf][m_local][ci]   (row stride 68 dwords)
  half_t g[2][128][40];     // [buf][ci][m_local]   (row stride 20 dwords)
};
union SMemU {
  StageBufs st;             // 37,888 B, used during the key loop
  half_t ybuf[8][16][136];  // 34,816 B, used after the key loop (disjoint)
};

__global__ __launch_bounds__(256) void k_attn(const float* __restrict__ x,
                                              const float* __restrict__ b_W,
                                              const float* __restrict__ bn_gamma,
                                              const float* __restrict__ bn_beta,
                                              const float* __restrict__ bn_mean,
                                              const float* __restrict__ bn_var,
                                              half_t* __restrict__ ws,
                                              float* __restrict__ out) {
  const int b    = blockIdx.y;
  const int tile = blockIdx.x;          // 0..31
  const int tid  = threadIdx.x;
  const int lane = tid & 31, wv = tid >> 5;
  const int lp = lane & 15, hg = lane >> 4;
  const int n0 = tile * 128 + wv * 16;  // this wave's query rows

  __shared__ SMemU sm;
  __shared__ half_t pbuf[8][16][40];    // per-wave P transpose scratch

  const half_t* thetaT = ws + OFF_THETA;
  const half_t* phiT   = ws + OFF_PHIT;
  const half_t* gbuf   = ws + OFF_G;
  const half_t* wW     = ws + OFF_WH + 3 * CI_ * C_;

  // stage one 32-key chunk (phi rows + g columns) into LDS buffer bf.
  // per thread: 2 phi segments + 2 g segments = 4 async b128 per wave issue.
  auto stage = [&](int chunk, int bf) {
    const int m0s = chunk * 32;
    const half_t* srcP = phiT + ((size_t)b * M_ + m0s) * CI_;  // 8KB contiguous
    const half_t* srcG = gbuf + (size_t)b * CI_ * M_ + m0s;    // 128 rows x 64B
    for (int s = tid; s < 512; s += 256) {
      const int j = s >> 4, q = s & 15;
      COPY16_TO_LDS(&sm.st.phi[bf][j][q * 8], srcP + j * CI_ + q * 8);
    }
    for (int s = tid; s < 512; s += 256) {
      const int ci = s >> 2, q = s & 3;
      COPY16_TO_LDS(&sm.st.g[bf][ci][q * 8], srcG + (size_t)ci * M_ + q * 8);
    }
  };

  // theta A-fragments are loop-invariant over keys: hoist all 4 K-steps.
  v16h AF[4];
  {
    const half_t* trow = thetaT + ((size_t)b * N_ + n0 + lp) * CI_;
#pragma unroll
    for (int kk = 0; kk < 4; ++kk) AF[kk] = load_a_frag(trow + kk * 32, hg);
  }

  const v8f vzero = {};
  float mrow[8], lrow[8];
  v8f yacc[8];
#pragma unroll
  for (int r = 0; r < 8; ++r) { mrow[r] = -3.0e38f; lrow[r] = 0.0f; }
#pragma unroll
  for (int t = 0; t < 8; ++t) yacc[t] = vzero;

  stage(0, 0);                          // prologue: fill buffer 0

  for (int mc = 0; mc < 32; ++mc) {     // key chunks of 32
    const int bf = mc & 1;
    if (mc + 1 < 32) {
      stage(mc + 1, bf ^ 1);            // overlap next chunk's DMA
      WAIT_ASYNC(4);                    // this wave's copies for chunk mc done
    } else {
      WAIT_ASYNC(0);
    }
    __syncthreads();                    // all waves' copies for chunk mc done

    // S = theta^T * phi  (two 16x16 output tiles, K = CI = 128) from LDS
    const half_t* prow0 = &sm.st.phi[bf][lp][16 * hg];
    const half_t* prow1 = &sm.st.phi[bf][16 + lp][16 * hg];
    v8f s0 = vzero, s1 = vzero;
#pragma unroll
    for (int kk = 0; kk < 4; ++kk) {
      s0 = wmma_f16(AF[kk], load_b_frag(prow0 + kk * 32), s0);
      s1 = wmma_f16(AF[kk], load_b_frag(prow1 + kk * 32), s1);
    }

    // online softmax over keys: row r+8*hg lives across the lane's 16-group
    float scale[8];
#pragma unroll
    for (int r = 0; r < 8; ++r) {
      float v = fmaxf(s0[r], s1[r]);
      v = fmaxf(v, __shfl_xor(v, 1, 32));
      v = fmaxf(v, __shfl_xor(v, 2, 32));
      v = fmaxf(v, __shfl_xor(v, 4, 32));
      v = fmaxf(v, __shfl_xor(v, 8, 32));
      float mnew = fmaxf(mrow[r], v);
      scale[r] = __expf(mrow[r] - mnew);
      float p0 = __expf(s0[r] - mnew);
      float p1 = __expf(s1[r] - mnew);
      mrow[r] = mnew;
      s0[r] = p0; s1[r] = p1;
      float rs = p0 + p1;
      rs += __shfl_xor(rs, 1, 32);
      rs += __shfl_xor(rs, 2, 32);
      rs += __shfl_xor(rs, 4, 32);
      rs += __shfl_xor(rs, 8, 32);
      lrow[r] = lrow[r] * scale[r] + rs;
    }
#pragma unroll
    for (int t = 0; t < 8; ++t) {
#pragma unroll
      for (int r = 0; r < 8; ++r) yacc[t][r] *= scale[r];
    }

    // transpose P (D layout -> A layout) through per-wave LDS
#pragma unroll
    for (int r = 0; r < 8; ++r) {
      pbuf[wv][r + 8 * hg][lp]      = (half_t)s0[r];
      pbuf[wv][r + 8 * hg][16 + lp] = (half_t)s1[r];
    }
    v16h PF = load_a_frag(&pbuf[wv][lp][0], hg);

    // y += P * g^T over all CI (8 output tiles, K = 32) from LDS
    const half_t* gr = &sm.st.g[bf][lp][16 * hg];
#pragma unroll
    for (int t = 0; t < 8; ++t)
      yacc[t] = wmma_f16(PF, load_b_frag(gr + t * 16 * 40), yacc[t]);

    __syncthreads();                    // done reading buf before it's refilled
  }

  // normalize and stash y tile [n_local][ci] in LDS (reuses staging storage;
  // temporally disjoint: the loop's final barrier has retired all readers)
  float linv[8];
#pragma unroll
  for (int r = 0; r < 8; ++r) linv[r] = 1.0f / lrow[r];
#pragma unroll
  for (int t = 0; t < 8; ++t)
#pragma unroll
    for (int r = 0; r < 8; ++r)
      sm.ybuf[wv][r + 8 * hg][t * 16 + lp] = (half_t)(yacc[t][r] * linv[r]);

  // W_y = w_W @ y, BN (eval), + residual; wave covers all 256 out channels
  for (int ct = 0; ct < 16; ++ct) {
    const int c0 = ct * 16;
    const half_t* wrow = wW + (c0 + lp) * CI_;
    v8f acc = vzero;
#pragma unroll
    for (int kk = 0; kk < 4; ++kk) {
      v16h A  = load_a_frag(wrow + kk * 32, hg);
      v16h Bf = load_b_frag(&sm.ybuf[wv][lp][kk * 32 + 16 * hg]);
      acc = wmma_f16(A, Bf, acc);
    }
#pragma unroll
    for (int r = 0; r < 8; ++r) {
      const int c = c0 + r + 8 * hg;
      const float inv = bn_gamma[c] * rsqrtf(bn_var[c] + 1e-5f);
      const float sh  = bn_beta[c] - bn_mean[c] * inv + b_W[c] * inv;
      const size_t o  = (size_t)(b * C_ + c) * N_ + (size_t)(n0 + lp);
      out[o] = acc[r] * inv + sh + x[o];
    }
  }
}

// ---------------- launch ----------------
extern "C" void kernel_launch(void* const* d_in, const int* in_sizes, int n_in,
                              void* d_out, int out_size, void* d_ws, size_t ws_size,
                              hipStream_t stream) {
  (void)in_sizes; (void)n_in; (void)out_size; (void)ws_size;
  const float* x       = (const float*)d_in[0];
  const float* w_theta = (const float*)d_in[1];
  const float* b_theta = (const float*)d_in[2];
  const float* w_phi   = (const float*)d_in[3];
  const float* b_phi   = (const float*)d_in[4];
  const float* w_g     = (const float*)d_in[5];
  const float* b_g     = (const float*)d_in[6];
  const float* w_W     = (const float*)d_in[7];
  const float* b_W     = (const float*)d_in[8];
  const float* bn_gamma= (const float*)d_in[9];
  const float* bn_beta = (const float*)d_in[10];
  const float* bn_mean = (const float*)d_in[11];
  const float* bn_var  = (const float*)d_in[12];

  half_t* ws  = (half_t*)d_ws;
  float*  out = (float*)d_out;

  k_cvt<<<dim3((CI_ * C_ + 255) / 256), dim3(256), 0, stream>>>(
      w_theta, w_phi, w_g, w_W, ws + OFF_WH);
  k_proj<<<dim3(64, B_), dim3(256), 0, stream>>>(x, b_theta, b_phi, b_g, ws);
  k_attn<<<dim3(32, B_), dim3(256), 0, stream>>>(
      x, b_W, bn_gamma, bn_beta, bn_mean, bn_var, ws, out);
}
